// UniMPGNN_23373212025515
// MI455X (gfx1250) — compile-verified, matrix operand order checked by reading, not run on hardware
//
#include <hip/hip_runtime.h>
#include <cstdint>

// ---------------------------------------------------------------------------
// UniMP GNN layer for MI455X (gfx1250, wave32, WMMA + async LDS DMA)
// ---------------------------------------------------------------------------

typedef __attribute__((ext_vector_type(16))) __bf16 v16bf;
typedef __attribute__((ext_vector_type(8)))  float  v8f;

#define D 128            // feature dim (in == out)
#define NUM_GRAPHS 16
#define WROW 136         // padded LDS row stride (bf16 elems) to rotate banks

// ---- float <-> monotone-unsigned encoding for atomicMax on floats ---------
__device__ __forceinline__ unsigned encodeF(float f) {
    unsigned u = __float_as_uint(f);
    return (u & 0x80000000u) ? ~u : (u | 0x80000000u);
}
__device__ __forceinline__ float decodeF(unsigned u) {
    unsigned b = (u & 0x80000000u) ? (u & 0x7fffffffu) : ~u;
    return __uint_as_float(b);
}
// encodeF(-inf) == 0x007FFFFF
#define ENC_NEG_INF 0x007FFFFFu

// ---------------------------------------------------------------------------
// Kernel 0 (one-time, 1 MB traffic): convert + transpose weights to bf16.
//   Wt[mat][n][k] = (bf16) W_mat[k][n]   -- contiguous, 4 x 128 x 128 bf16.
//   This 128 KB buffer stays L2-resident; every proj workgroup DMAs it.
// ---------------------------------------------------------------------------
__global__ void prepW_kernel(const float* __restrict__ W1, const float* __restrict__ W2,
                             const float* __restrict__ W3, const float* __restrict__ W4,
                             __bf16* __restrict__ Wt)
{
    const float* Ws[4] = {W1, W2, W3, W4};
    int o = blockIdx.x * blockDim.x + threadIdx.x;
    if (o >= 4 * D * D) return;
    int mat = o >> 14;           // / 16384
    int n   = (o >> 7) & (D - 1);
    int k   = o & (D - 1);
    Wt[o] = (__bf16)Ws[mat][k * D + n];
}

// ---------------------------------------------------------------------------
// Kernel 1: fused projections  x1..x4 = x @ W{1..4} + b{1..4}   (WMMA bf16)
//   One workgroup = 128 rows of x; 8 waves, each wave owns a 16-row stripe
//   and produces 16x128 for each of the 4 weight matrices.
//   Weights staged into LDS via GLOBAL_LOAD_ASYNC_TO_LDS_B128 (ASYNCcnt),
//   per-lane LDS addresses re-insert the 272B padded row stride.
// ---------------------------------------------------------------------------
__global__ __launch_bounds__(256) void proj_kernel(
    const float* __restrict__ x,
    const __bf16* __restrict__ Wt,               // [4][128][128] bf16 (n-major)
    const float* __restrict__ b1, const float* __restrict__ b2,
    const float* __restrict__ b3, const float* __restrict__ b4,
    float* __restrict__ o1, float* __restrict__ o2,
    float* __restrict__ o3, float* __restrict__ o4, int N)
{
    extern __shared__ char smem_raw[];
    __bf16* sW = (__bf16*)smem_raw;              // 4 x [128 cols][WROW k] padded

    const float* bs[4]   = {b1, b2, b3, b4};
    float*       outs[4] = {o1, o2, o3, o4};

    const int tid = threadIdx.x;

    // ---- async DMA: global (bf16, contiguous) -> LDS (padded rows) --------
    // 16B granules: granule g -> global byte g*16;
    //   mat = g>>11, row = (g>>4)&127, chunk = g&15
    //   LDS byte = mat*128*2*WROW + row*2*WROW + chunk*16
    {
        unsigned g = (unsigned)tid;
        #pragma unroll 1
        for (int it = 0; it < (4 * D * 16) / 256; ++it, g += 256) {
            unsigned mat      = g >> 11;
            unsigned rem      = g & 2047u;
            unsigned row      = rem >> 4;
            unsigned c16      = rem & 15u;
            unsigned lds_byte = mat * (D * 2 * WROW) + row * (2 * WROW) + c16 * 16;
            unsigned goff     = g * 16;
            asm volatile("global_load_async_to_lds_b128 %0, %1, %2"
                         :: "v"(lds_byte), "v"(goff), "s"(Wt)
                         : "memory");
        }
        asm volatile("s_wait_asynccnt 0x0" ::: "memory");
    }
    __syncthreads();

    const int wave = tid >> 5;
    const int lane = tid & 31;
    const int rowbase = blockIdx.x * 128 + wave * 16;
    if (rowbase >= N) return;                    // N % 16 == 0 -> full tiles only

    const int r     = rowbase + (lane & 15);     // A-matrix row for this lane
    const int khalf = (lane >> 4) * 8;           // lanes 16-31 hold K+8 block

    // Load A fragments (4 K-chunks of 32), converting f32 -> bf16 on the fly.
    // 16-bit A 16x32 layout: elems 0-7 -> K = kc*32 + khalf + e
    //                        elems 8-15 -> K = kc*32 + 16 + khalf + (e-8)
    v16bf A[4];
    const float* xrow = x + (size_t)r * D;
    #pragma unroll
    for (int kc = 0; kc < 4; ++kc) {
        const int k0 = kc * 32 + khalf;
        #pragma unroll
        for (int e = 0; e < 8; ++e) A[kc][e]     = (__bf16)xrow[k0 + e];
        #pragma unroll
        for (int e = 0; e < 8; ++e) A[kc][8 + e] = (__bf16)xrow[k0 + 16 + e];
    }

    const int n_lane = lane & 15;
    const int rhalf  = (lane >> 4) * 8;          // C/D row group for this lane

    #pragma unroll
    for (int mat = 0; mat < 4; ++mat) {
        float*       outp = outs[mat];
        const float* bias = bs[mat];
        const __bf16* wmat = &sW[mat * D * WROW];
        #pragma unroll
        for (int ct = 0; ct < 8; ++ct) {         // 8 column tiles of 16
            const int n  = ct * 16 + n_lane;
            const float bv = bias[n];
            v8f c;
            #pragma unroll
            for (int v = 0; v < 8; ++v) c[v] = bv;  // bias-initialized accumulator

            const __bf16* wcol = wmat + n * WROW;   // W^T row = output column n
            #pragma unroll
            for (int kc = 0; kc < 4; ++kc) {
                v16bf B;                            // B 32x16: same K pattern as A
                const int k0 = kc * 32 + khalf;
                #pragma unroll
                for (int e = 0; e < 8; ++e) B[e]     = wcol[k0 + e];
                #pragma unroll
                for (int e = 0; e < 8; ++e) B[8 + e] = wcol[k0 + 16 + e];
                c = __builtin_amdgcn_wmma_f32_16x16x32_bf16(
                        false, A[kc], false, B, (short)0, c, false, false);
            }
            // D layout: vgpr v, lane l -> row = v + 8*(l>=16), col = l&15
            #pragma unroll
            for (int v = 0; v < 8; ++v)
                outp[(size_t)(rowbase + rhalf + v) * D + n] = c[v];
        }
    }
}

// ---------------------------------------------------------------------------
// Kernel 2: init segment-max (encoded -inf), segment-sum z, pooled, counts
// ---------------------------------------------------------------------------
__global__ void init_kernel(unsigned* __restrict__ mEnc, float* __restrict__ z,
                            float* __restrict__ pooled, float* __restrict__ counts,
                            int N)
{
    int i = blockIdx.x * blockDim.x + threadIdx.x;
    if (i < N) { mEnc[i] = ENC_NEG_INF; z[i] = 0.0f; }
    if (i < NUM_GRAPHS * D) pooled[i] = 0.0f;
    if (i < NUM_GRAPHS)     counts[i] = 0.0f;
}

// ---------------------------------------------------------------------------
// Kernel 3: per-node rank-1 edge-term dots  p[i]=<x3[i],W5>, q[i]=<x3[i],b5>
// ---------------------------------------------------------------------------
__global__ void pq_kernel(const float* __restrict__ x3,
                          const float* __restrict__ W5, const float* __restrict__ b5,
                          float* __restrict__ p, float* __restrict__ q, int N)
{
    int gid  = blockIdx.x * blockDim.x + threadIdx.x;
    int node = gid >> 5;
    int lane = gid & 31;
    if (node >= N) return;
    const float* a = x3 + (size_t)node * D;
    float sp = 0.0f, sq = 0.0f;
    #pragma unroll
    for (int c = 0; c < D; c += 32) {
        float av = a[c + lane];
        sp += av * W5[c + lane];
        sq += av * b5[c + lane];
    }
    #pragma unroll
    for (int off = 16; off > 0; off >>= 1) {
        sp += __shfl_xor(sp, off, 32);
        sq += __shfl_xor(sq, off, 32);
    }
    if (lane == 0) { p[node] = sp; q[node] = sq; }
}

// ---------------------------------------------------------------------------
// Kernel 4: per-edge attention score + segment max (one wave per edge)
// ---------------------------------------------------------------------------
__global__ void score_kernel(const long long* __restrict__ ei,
                             const float* __restrict__ ea,
                             const float* __restrict__ x3, const float* __restrict__ x4,
                             const float* __restrict__ p,  const float* __restrict__ q,
                             float* __restrict__ scores, unsigned* __restrict__ mEnc,
                             int E)
{
    int gid  = blockIdx.x * blockDim.x + threadIdx.x;
    int e    = gid >> 5;
    int lane = gid & 31;
    if (e >= E) return;
    int src = (int)ei[e];
    int dst = (int)ei[(size_t)E + e];
    const float* a = x3 + (size_t)dst * D;
    const float* b = x4 + (size_t)src * D;
    float s = 0.0f;
    #pragma unroll
    for (int c = 0; c < D; c += 32) s += a[c + lane] * b[c + lane];
    #pragma unroll
    for (int off = 16; off > 0; off >>= 1) s += __shfl_xor(s, off, 32);
    if (lane == 0) {
        float sc = (s + ea[e] * p[dst] + q[dst]) * 0.08838834764831845f; // 1/sqrt(128)
        scores[e] = sc;
        atomicMax(mEnc + dst, encodeF(sc));
    }
}

// ---------------------------------------------------------------------------
// Kernel 5: exp(score - m[dst]) and segment sum z (one thread per edge)
// ---------------------------------------------------------------------------
__global__ void exp_kernel(const long long* __restrict__ ei,
                           const float* __restrict__ scores,
                           const unsigned* __restrict__ mEnc,
                           float* __restrict__ expw, float* __restrict__ z, int E)
{
    int e = blockIdx.x * blockDim.x + threadIdx.x;
    if (e >= E) return;
    int dst = (int)ei[(size_t)E + e];
    float v = __expf(scores[e] - decodeF(mEnc[dst]));
    expw[e] = v;
    atomicAdd(z + dst, v);
}

// ---------------------------------------------------------------------------
// Kernel 6: weighted scatter-add  x1[dst] += alpha * x2[src]  (wave per edge)
// ---------------------------------------------------------------------------
__global__ void aggr_kernel(const long long* __restrict__ ei,
                            const float* __restrict__ expw, const float* __restrict__ z,
                            const float* __restrict__ x2, float* __restrict__ x1,
                            int E)
{
    int gid  = blockIdx.x * blockDim.x + threadIdx.x;
    int e    = gid >> 5;
    int lane = gid & 31;
    if (e >= E) return;
    int src = (int)ei[e];
    int dst = (int)ei[(size_t)E + e];
    float alpha = expw[e] / z[dst];
    const float4 v = reinterpret_cast<const float4*>(x2 + (size_t)src * D)[lane];
    float* o = x1 + (size_t)dst * D + lane * 4;
    atomicAdd(o + 0, alpha * v.x);
    atomicAdd(o + 1, alpha * v.y);
    atomicAdd(o + 2, alpha * v.z);
    atomicAdd(o + 3, alpha * v.w);
}

// ---------------------------------------------------------------------------
// Kernel 7: global pool accumulation (wave per node)
// ---------------------------------------------------------------------------
__global__ void pool_kernel(const float* __restrict__ x1,
                            const long long* __restrict__ batch,
                            float* __restrict__ pooled, float* __restrict__ counts,
                            int N)
{
    int gid  = blockIdx.x * blockDim.x + threadIdx.x;
    int node = gid >> 5;
    int lane = gid & 31;
    if (node >= N) return;
    int g = (int)batch[node];
    const float4 v = reinterpret_cast<const float4*>(x1 + (size_t)node * D)[lane];
    float* pp = pooled + (size_t)g * D + lane * 4;
    atomicAdd(pp + 0, v.x);
    atomicAdd(pp + 1, v.y);
    atomicAdd(pp + 2, v.z);
    atomicAdd(pp + 3, v.w);
    if (lane == 0) atomicAdd(counts + g, 1.0f);
}

// ---------------------------------------------------------------------------
// Kernel 8: mean divide into d_out
// ---------------------------------------------------------------------------
__global__ void final_kernel(const float* __restrict__ pooled,
                             const float* __restrict__ counts,
                             float* __restrict__ out)
{
    int i = blockIdx.x * blockDim.x + threadIdx.x;
    if (i < NUM_GRAPHS * D) out[i] = pooled[i] / fmaxf(counts[i >> 7], 1.0f);
}

// ---------------------------------------------------------------------------
extern "C" void kernel_launch(void* const* d_in, const int* in_sizes, int n_in,
                              void* d_out, int out_size, void* d_ws, size_t ws_size,
                              hipStream_t stream)
{
    const float*     x     = (const float*)d_in[0];
    const long long* ei    = (const long long*)d_in[1];   // int64 [2, E]
    const float*     ea    = (const float*)d_in[2];
    const long long* batch = (const long long*)d_in[3];
    const float* W1 = (const float*)d_in[4];  const float* b1 = (const float*)d_in[5];
    const float* W2 = (const float*)d_in[6];  const float* b2 = (const float*)d_in[7];
    const float* W3 = (const float*)d_in[8];  const float* b3 = (const float*)d_in[9];
    const float* W4 = (const float*)d_in[10]; const float* b4 = (const float*)d_in[11];
    const float* W5 = (const float*)d_in[12]; const float* b5 = (const float*)d_in[13];

    const int N = in_sizes[0] / D;       // 100000
    const int E = in_sizes[2];           // edge_attr is [E,1] -> 600000

    // ---- workspace layout (all 4-byte elements; 16B-aligned sections) ----
    float* ws = (float*)d_ws;
    const size_t NB = (size_t)N * D;
    float*    x1     = ws;               // node_out accumulator (x@W1+b1 + aggr)
    float*    x2     = ws + NB;
    float*    x3     = ws + 2 * NB;
    float*    x4     = ws + 3 * NB;
    float*    p      = ws + 4 * NB;
    float*    q      = p + N;
    unsigned* mEnc   = (unsigned*)(q + N);
    float*    z      = (float*)(mEnc + N);
    float*    scores = z + N;
    float*    expw   = scores + E;
    float*    pooled = expw + E;
    float*    counts = pooled + NUM_GRAPHS * D;
    __bf16*   Wt     = (__bf16*)(counts + NUM_GRAPHS);  // 4*128*128 bf16 (128 KB)
    float*    out    = (float*)d_out;

    // 0) one-time weight convert+transpose to bf16 (L2-resident afterwards)
    prepW_kernel<<<(4 * D * D + 255) / 256, 256, 0, stream>>>(W1, W2, W3, W4, Wt);

    // 1) fused WMMA projections (dynamic LDS: 4 padded bf16 weight mats,
    //    filled by GLOBAL_LOAD_ASYNC_TO_LDS_B128 DMA)
    const size_t ldsBytes = 4u * D * WROW * sizeof(__bf16);   // 139264 B
    proj_kernel<<<(N + 127) / 128, 256, ldsBytes, stream>>>(
        x, Wt, b1, b2, b3, b4, x1, x2, x3, x4, N);

    // 2) init reductions
    init_kernel<<<(N + 255) / 256, 256, 0, stream>>>(mEnc, z, pooled, counts, N);

    // 3) per-node rank-1 dots
    pq_kernel<<<(N * 32 + 255) / 256, 256, 0, stream>>>(x3, W5, b5, p, q, N);

    // 4) per-edge scores + segment max
    score_kernel<<<(int)(((long long)E * 32 + 255) / 256), 256, 0, stream>>>(
        ei, ea, x3, x4, p, q, scores, mEnc, E);

    // 5) exp + segment sum
    exp_kernel<<<(E + 255) / 256, 256, 0, stream>>>(ei, scores, mEnc, expw, z, E);

    // 6) weighted scatter-add into x1
    aggr_kernel<<<(int)(((long long)E * 32 + 255) / 256), 256, 0, stream>>>(
        ei, expw, z, x2, x1, E);

    // 7) global mean pool: accumulate
    pool_kernel<<<(N * 32 + 255) / 256, 256, 0, stream>>>(x1, batch, pooled, counts, N);

    // 8) divide -> output [16, 128]
    final_kernel<<<(NUM_GRAPHS * D + 255) / 256, 256, 0, stream>>>(pooled, counts, out);
}
